// MultiHeadSelfAttention_35682588295575
// MI455X (gfx1250) — matrix operand order repaired and checked by reading, not compiled, not run on hardware
//
#include <hip/hip_runtime.h>
#include <hip/hip_fp16.h>
#include <stdint.h>

typedef __attribute__((ext_vector_type(16))) _Float16 v16h;
typedef __attribute__((ext_vector_type(8)))  _Float16 h8;
typedef __attribute__((ext_vector_type(8)))  float    v8f;
typedef __attribute__((ext_vector_type(4))) unsigned int v4u;
typedef __attribute__((ext_vector_type(8))) int v8i;
typedef __attribute__((ext_vector_type(4))) int v4i;

namespace {
constexpr int BATCH = 4;
constexpr int SEQ   = 2048;
constexpr int HEADS = 8;
constexpr int DHEAD = 64;              // DK == DV == 64
constexpr int NQK   = HEADS * DHEAD;   // 512

__device__ __forceinline__ v16h hcat(h8 lo, h8 hi) {
    return __builtin_shufflevector(lo, hi, 0,1,2,3,4,5,6,7,8,9,10,11,12,13,14,15);
}
__device__ __forceinline__ h8 ld8(const _Float16* p) { return *(const h8*)p; }
__device__ __forceinline__ v16h ld16(const _Float16* p) {
    const h8* t = (const h8*)p;
    return hcat(t[0], t[1]);
}
__device__ __forceinline__ v8f wmma(v16h a, v16h b, v8f c) {
    // D = A(16x32,f16) * B(32x16,f16) + C(16x16,f32)
    return __builtin_amdgcn_wmma_f32_16x16x32_f16(false, a, false, b, (short)0, c,
                                                  false, false);
}

// ---------------------------------------------------------------------------
// TDM: async 2-D tile load Global -> LDS (CDNA5 ISA).
// Elements are f16 (data_size code 1).  X (dim0) is the contiguous dimension.
// D# group0: [1:0]=count=1 | [63:32]=lds_addr | [120:64]=global_addr | type=2
// D# group1: [17:16]=data_size | tensor_dim0/1, tile_dim0/1 | tensor_dim0_stride
// groups 2/3 zero (2-D tile; tile_dim2/3 = 0 -> unused).
// ---------------------------------------------------------------------------
__device__ __forceinline__ void tdm_load_tile_2d(
    uint32_t lds_off, const void* gptr,
    uint32_t tensor_d0, uint32_t tensor_d1,
    uint32_t tile_d0, uint32_t tile_d1,
    uint64_t row_stride_elems)
{
    const uint64_t ga = (uint64_t)(uintptr_t)gptr;
    v4u g0 = { 1u,                                  // count=1, user descriptor
               lds_off,                             // LDS byte address
               (uint32_t)ga,                        // global addr [31:0]
               (uint32_t)(ga >> 32) | (2u << 30) }; // addr [56:32] | type=2
    v8i g1 = {
        (int)(1u << 16),                                          // data_size=1 (2B)
        (int)((tensor_d0 & 0xffffu) << 16),                       // tensor_dim0[15:0]
        (int)((tensor_d0 >> 16) | ((tensor_d1 & 0xffffu) << 16)), // td0 hi | td1 lo
        (int)((tensor_d1 >> 16) | (tile_d0 << 16)),               // td1 hi | tile_dim0
        (int)(tile_d1 & 0xffffu),                                 // tile_dim1 | tile_dim2=0
        (int)(uint32_t)row_stride_elems,                          // dim0_stride[31:0]
        (int)(uint32_t)((row_stride_elems >> 32) & 0xffffu),      // dim0_stride[47:32]
        0 };
    v4i z4 = {0, 0, 0, 0};
#if __clang_major__ >= 23
    v8i z8 = {0, 0, 0, 0, 0, 0, 0, 0};
    __builtin_amdgcn_tensor_load_to_lds(g0, g1, z4, z4, z8, 0);
#else
    __builtin_amdgcn_tensor_load_to_lds(g0, g1, z4, z4, 0);
#endif
}
} // namespace

// ---------------------------------------------------------------------------
// Kernel 1: QKV projections. One wave = one 16x16 output tile of q/k/v.
//   q,k stored f16 [b,h,l,d]; v stored f16 transposed [b,h,d,l].
// ---------------------------------------------------------------------------
__global__ __launch_bounds__(128) void qkv_proj_wmma(
    const float* __restrict__ x,
    const float* __restrict__ wq, const float* __restrict__ bq,
    const float* __restrict__ wk, const float* __restrict__ bk,
    const float* __restrict__ wv, const float* __restrict__ bv,
    _Float16* __restrict__ qh, _Float16* __restrict__ kh, _Float16* __restrict__ vt)
{
    const int lane = threadIdx.x & 31;
    const int half = lane >> 4;
    const int nl   = lane & 15;
    int wave = (blockIdx.x * blockDim.x + threadIdx.x) >> 5;

    constexpr int NT = NQK / 16;            // 32 column tiles
    constexpr int MT = (BATCH * SEQ) / 16;  // 512 row tiles
    const int proj  = wave / (MT * NT);     // 0=q 1=k 2=v
    wave -= proj * MT * NT;
    const int mtile = wave / NT;
    const int ntile = wave - mtile * NT;

    const float* W    = (proj == 0) ? wq : (proj == 1) ? wk : wv;
    const float* bias = (proj == 0) ? bq : (proj == 1) ? bk : bv;

    const float* xr = x + (size_t)(mtile * 16 + nl) * DHEAD;
    v16h a0, a1;
#pragma unroll
    for (int j = 0; j < 8; ++j) {
        a0[j]     = (_Float16)xr[half * 16 + j];
        a0[j + 8] = (_Float16)xr[half * 16 + 8 + j];
        a1[j]     = (_Float16)xr[32 + half * 16 + j];
        a1[j + 8] = (_Float16)xr[40 + half * 16 + j];
    }
    const int col = ntile * 16 + nl;
    v16h b0, b1;
#pragma unroll
    for (int j = 0; j < 16; ++j) {
        b0[j] = (_Float16)W[(size_t)(half * 16 + j) * NQK + col];
        b1[j] = (_Float16)W[(size_t)(32 + half * 16 + j) * NQK + col];
    }
    v8f c = {};
    c = wmma(a0, b0, c);
    c = wmma(a1, b1, c);

    const float bia = bias[col];
    const int hh = col >> 6;
    const int dd = col & 63;
#pragma unroll
    for (int r = 0; r < 8; ++r) {
        const int grow = mtile * 16 + half * 8 + r;
        const int bb = grow >> 11;
        const int ll = grow & (SEQ - 1);
        const _Float16 v = (_Float16)(c[r] + bia);
        if (proj == 0)
            qh[((size_t)(bb * HEADS + hh) * SEQ + ll) * DHEAD + dd] = v;
        else if (proj == 1)
            kh[((size_t)(bb * HEADS + hh) * SEQ + ll) * DHEAD + dd] = v;
        else
            vt[((size_t)(bb * HEADS + hh) * DHEAD + dd) * SEQ + ll] = v;
    }
}

// ---------------------------------------------------------------------------
// Kernel 2: flash attention.  One block = one (b,h), 4 waves = 4 q-tiles.
//   TDM double-buffers the 32-key K tile (32x64) and V^T tile (64x32) in LDS,
//   shared by all 4 waves; 4 WMMAs for S, online softmax, P through LDS
//   (C/D->A layout transpose), 4 WMMAs for O += P*V per block.
//   Reference quirk: scores are MULTIPLIED by sqrt(dk)=8.
// ---------------------------------------------------------------------------
__global__ __launch_bounds__(128) void attn_wmma(
    const _Float16* __restrict__ qh, const _Float16* __restrict__ kh,
    const _Float16* __restrict__ vt, _Float16* __restrict__ ctx)
{
    __shared__ _Float16 Kbuf[2][32 * DHEAD];   // [key][d]   8 KB
    __shared__ _Float16 Vbuf[2][DHEAD * 32];   // [d][key]   8 KB
    __shared__ _Float16 Pstage[4][16 * 32];    // per-wave   4 KB

    const int wid  = threadIdx.x >> 5;
    const int lane = threadIdx.x & 31;
    const int half = lane >> 4;
    const int nl   = lane & 15;

    const int bh    = blockIdx.x >> 5;              // 0..31  (b*H + h)
    const int qtile = (blockIdx.x & 31) * 4 + wid;  // 0..127

    const _Float16* qp = qh + ((size_t)bh * SEQ + qtile * 16) * DHEAD;
    const _Float16* kp = kh + (size_t)bh * SEQ * DHEAD;
    const _Float16* vp = vt + (size_t)bh * DHEAD * SEQ;

    // Issue TDM loads of key-block 0 while Q fragments are fetched.
    if (wid == 0) {
        tdm_load_tile_2d((uint32_t)(uintptr_t)&Kbuf[0][0], kp,
                         DHEAD, SEQ, DHEAD, 32, DHEAD);
        tdm_load_tile_2d((uint32_t)(uintptr_t)&Vbuf[0][0], vp,
                         SEQ, DHEAD, 32, DHEAD, SEQ);
    }

    // Persistent Q A-fragments (16x64 as two 16x32 tiles)
    const _Float16* qr = qp + (size_t)nl * DHEAD;
    const v16h aq0 = hcat(ld8(qr + half * 16),      ld8(qr + 8 + half * 16));
    const v16h aq1 = hcat(ld8(qr + 32 + half * 16), ld8(qr + 40 + half * 16));

    v8f o0 = {}, o1 = {}, o2 = {}, o3 = {};
    float mrun[8], lrun[8];   // reg r <-> row half*8 + r (matches C/D layout)
#pragma unroll
    for (int r = 0; r < 8; ++r) { mrun[r] = -3.0e38f; lrun[r] = 0.0f; }

    _Float16* Pl = Pstage[wid];

    for (int kb = 0, it = 0; kb < SEQ; kb += 32, ++it) {
        const int cur = it & 1;
        if (wid == 0) {
            if (kb + 32 < SEQ) {
                // prefetch next block into the other buffer
                tdm_load_tile_2d((uint32_t)(uintptr_t)&Kbuf[cur ^ 1][0],
                                 kp + (size_t)(kb + 32) * DHEAD,
                                 DHEAD, SEQ, DHEAD, 32, DHEAD);
                tdm_load_tile_2d((uint32_t)(uintptr_t)&Vbuf[cur ^ 1][0],
                                 vp + (size_t)(kb + 32),
                                 SEQ, DHEAD, 32, DHEAD, SEQ);
                // in-order completion: <=2 outstanding => current block landed
                __builtin_amdgcn_s_wait_tensorcnt(2);
            } else {
                __builtin_amdgcn_s_wait_tensorcnt(0);
            }
        }
        __syncthreads();   // current tiles visible to all 4 waves

        const _Float16* Kl = Kbuf[cur];
        const _Float16* Vl = Vbuf[cur];

        // ---- S (16 rows x 32 keys) = Q * K^T, K rows from LDS ----
        v8f s0 = {}, s1 = {};
        s0 = wmma(aq0, ld16(Kl + (size_t)nl * DHEAD + half * 16), s0);
        s0 = wmma(aq1, ld16(Kl + (size_t)nl * DHEAD + 32 + half * 16), s0);
        s1 = wmma(aq0, ld16(Kl + (size_t)(16 + nl) * DHEAD + half * 16), s1);
        s1 = wmma(aq1, ld16(Kl + (size_t)(16 + nl) * DHEAD + 32 + half * 16), s1);

        // ---- scale (multiply by 8 per reference) + block row max ----
        float bm[8];
#pragma unroll
        for (int r = 0; r < 8; ++r) {
            s0[r] *= 8.0f; s1[r] *= 8.0f;
            bm[r] = fmaxf(s0[r], s1[r]);
        }
#pragma unroll
        for (int xm = 1; xm < 16; xm <<= 1)
#pragma unroll
            for (int r = 0; r < 8; ++r)
                bm[r] = fmaxf(bm[r], __shfl_xor(bm[r], xm, 32));

        // ---- online softmax ----
        float fac[8], ps[8];
#pragma unroll
        for (int r = 0; r < 8; ++r) {
            const float mn = fmaxf(mrun[r], bm[r]);
            fac[r] = __expf(mrun[r] - mn);
            mrun[r] = mn;
            s0[r] = __expf(s0[r] - mn);
            s1[r] = __expf(s1[r] - mn);
            ps[r] = s0[r] + s1[r];
        }
#pragma unroll
        for (int xm = 1; xm < 16; xm <<= 1)
#pragma unroll
            for (int r = 0; r < 8; ++r)
                ps[r] += __shfl_xor(ps[r], xm, 32);
#pragma unroll
        for (int r = 0; r < 8; ++r) {
            lrun[r] = lrun[r] * fac[r] + ps[r];
            o0[r] *= fac[r]; o1[r] *= fac[r]; o2[r] *= fac[r]; o3[r] *= fac[r];
        }

        // ---- P: C/D layout -> A layout via per-wave LDS tile ----
        asm volatile("" ::: "memory");
#pragma unroll
        for (int r = 0; r < 8; ++r) {
            Pl[(half * 8 + r) * 32 + nl]      = (_Float16)s0[r];
            Pl[(half * 8 + r) * 32 + 16 + nl] = (_Float16)s1[r];
        }
        asm volatile("" ::: "memory");      // same-wave DS ops are in-order
        const _Float16* prow = Pl + nl * 32;
        const v16h ap = hcat(ld8(prow + half * 16), ld8(prow + 8 + half * 16));

        // ---- O += P * V, V^T rows from LDS (contiguous along keys) ----
        o0 = wmma(ap, ld16(Vl + (size_t)(nl)      * 32 + half * 16), o0);
        o1 = wmma(ap, ld16(Vl + (size_t)(16 + nl) * 32 + half * 16), o1);
        o2 = wmma(ap, ld16(Vl + (size_t)(32 + nl) * 32 + half * 16), o2);
        o3 = wmma(ap, ld16(Vl + (size_t)(48 + nl) * 32 + half * 16), o3);

        __syncthreads();   // everyone done with buffer `cur` before reuse
    }

    // ---- normalize and store ctx in reference layout [b, l, dv*H + h] ----
    const int hh = bh & (HEADS - 1);
    const int bb = bh >> 3;
    _Float16* cb = ctx + ((size_t)bb * SEQ + qtile * 16) * NQK;
#pragma unroll
    for (int r = 0; r < 8; ++r) {
        const float inv = 1.0f / lrun[r];
        _Float16* rowp = cb + (size_t)(half * 8 + r) * NQK + hh;
        rowp[(0 * 16 + nl) * HEADS] = (_Float16)(o0[r] * inv);
        rowp[(1 * 16 + nl) * HEADS] = (_Float16)(o1[r] * inv);
        rowp[(2 * 16 + nl) * HEADS] = (_Float16)(o2[r] * inv);
        rowp[(3 * 16 + nl) * HEADS] = (_Float16)(o3[r] * inv);
    }
}

// ---------------------------------------------------------------------------
// Kernel 3: out = ctx[8192,512] @ wo[512,64] + bo. One wave = one 16x16 tile.
// ---------------------------------------------------------------------------
__global__ __launch_bounds__(128) void out_proj_wmma(
    const _Float16* __restrict__ ctx, const float* __restrict__ wo,
    const float* __restrict__ bo, float* __restrict__ out)
{
    const int lane = threadIdx.x & 31;
    const int half = lane >> 4;
    const int nl   = lane & 15;
    const int wave = (blockIdx.x * blockDim.x + threadIdx.x) >> 5;
    const int mtile = wave >> 2;
    const int ntile = wave & 3;

    const _Float16* arow = ctx + (size_t)(mtile * 16 + nl) * NQK;
    const int col = ntile * 16 + nl;
    v8f c = {};
#pragma unroll
    for (int ks = 0; ks < NQK; ks += 32) {
        const v16h a = hcat(ld8(arow + ks + half * 16), ld8(arow + ks + 8 + half * 16));
        v16h b;
#pragma unroll
        for (int j = 0; j < 16; ++j)
            b[j] = (_Float16)wo[(size_t)(ks + half * 16 + j) * DHEAD + col];
        c = wmma(a, b, c);
    }
    const float bia = bo[col];
#pragma unroll
    for (int r = 0; r < 8; ++r)
        out[(size_t)(mtile * 16 + half * 8 + r) * DHEAD + col] = c[r] + bia;
}

// ---------------------------------------------------------------------------
extern "C" void kernel_launch(void* const* d_in, const int* in_sizes, int n_in,
                              void* d_out, int out_size, void* d_ws, size_t ws_size,
                              hipStream_t stream) {
    (void)in_sizes; (void)n_in; (void)out_size; (void)ws_size;
    const float* x  = (const float*)d_in[0];
    const float* wq = (const float*)d_in[1];
    const float* bq = (const float*)d_in[2];
    const float* wk = (const float*)d_in[3];
    const float* bk = (const float*)d_in[4];
    const float* wv = (const float*)d_in[5];
    const float* bv = (const float*)d_in[6];
    const float* wo = (const float*)d_in[7];
    const float* bo = (const float*)d_in[8];
    float* out = (float*)d_out;

    // Scratch (f16): q | k | vT | ctx  -> 4 x 4,194,304 halves = 32 MB
    _Float16* ws = (_Float16*)d_ws;
    const size_t SEG = (size_t)BATCH * HEADS * SEQ * DHEAD;
    _Float16* qh  = ws;
    _Float16* kh  = ws + SEG;
    _Float16* vt  = ws + 2 * SEG;
    _Float16* ctx = ws + 3 * SEG;

    qkv_proj_wmma<<<dim3(12288), dim3(128), 0, stream>>>(x, wq, bq, wk, bk, wv, bv,
                                                         qh, kh, vt);
    attn_wmma<<<dim3(1024), dim3(128), 0, stream>>>(qh, kh, vt, ctx);
    out_proj_wmma<<<dim3(512), dim3(128), 0, stream>>>(ctx, wo, bo, out);
}
